// LiquidMambaKAN_33328946217764
// MI455X (gfx1250) — matrix operand order repaired and checked by reading, not compiled, not run on hardware
//
#include <hip/hip_runtime.h>
#include <hip/hip_bf16.h>
#include <cstdint>
#include <cstddef>

// ---------------- model constants ----------------
#define BATCH 8
#define SEQ   512
#define DIN   64
#define DM    512
#define NL    4
#define DI    1024   // D_INNER
#define DST   16     // D_STATE
#define DTR   32     // DT_RANK
#define TOK   (BATCH*SEQ)   // 4096

typedef __attribute__((ext_vector_type(16))) __bf16 v16bf;
typedef __attribute__((ext_vector_type(8)))  float  v8f;

struct __attribute__((aligned(16))) F4  { float x, y, z, w; };
struct __attribute__((aligned(16))) BF8 { __bf16 h[8]; };

__device__ __forceinline__ int imin(int a, int b) { return a < b ? a : b; }

// ---------------- WMMA fragment helpers (unconditional; callers clamp addresses) ----------------
// A fragment (16x32, bf16) from f32 row-major source.
// lane&15 = row, kw=(lane>>4)*8; elems 0..7 -> K=kw..kw+7, 8..15 -> K=kw+16..kw+23.
__device__ __forceinline__ v16bf load_a_frag_f32(const float* rowp, int kw) {
  F4 x0 = *(const F4*)(rowp + kw);
  F4 x1 = *(const F4*)(rowp + kw + 4);
  F4 y0 = *(const F4*)(rowp + kw + 16);
  F4 y1 = *(const F4*)(rowp + kw + 20);
  v16bf f;
  f[0]=(__bf16)x0.x;  f[1]=(__bf16)x0.y;  f[2]=(__bf16)x0.z;  f[3]=(__bf16)x0.w;
  f[4]=(__bf16)x1.x;  f[5]=(__bf16)x1.y;  f[6]=(__bf16)x1.z;  f[7]=(__bf16)x1.w;
  f[8]=(__bf16)y0.x;  f[9]=(__bf16)y0.y;  f[10]=(__bf16)y0.z; f[11]=(__bf16)y0.w;
  f[12]=(__bf16)y1.x; f[13]=(__bf16)y1.y; f[14]=(__bf16)y1.z; f[15]=(__bf16)y1.w;
  return f;
}

// A fragment from a bf16 row (LDS h mirror).
__device__ __forceinline__ v16bf load_a_frag_bf16(const __bf16* rowp, int kw) {
  BF8 u0 = *(const BF8*)(rowp + kw);
  BF8 u1 = *(const BF8*)(rowp + kw + 16);
  v16bf f;
#pragma unroll
  for (int j = 0; j < 8; ++j) { f[j] = u0.h[j]; f[8 + j] = u1.h[j]; }
  return f;
}

// B fragment (32x16, bf16) from (N,K) row-major bf16 weights.
// lane&15 = column n; kw=(lane>>4)*8 (=kh/2); elems 0..7 -> K=kw..kw+7, 8..15 -> K=kw+16..kw+23.
__device__ __forceinline__ v16bf load_b_frag(const __bf16* rowp, int kh) {
  BF8 u0 = *(const BF8*)(rowp + (kh >> 1));
  BF8 u1 = *(const BF8*)(rowp + (kh >> 1) + 16);
  v16bf f;
#pragma unroll
  for (int j = 0; j < 8; ++j) { f[j] = u0.h[j]; f[8 + j] = u1.h[j]; }
  return f;
}

// ---------------- generic WMMA GEMM: C(MxN) = A(MxK,f32) * B(NxK,bf16)^T [+ Cadd] ----------------
// block = 256 threads (8 waves); wave -> 32x32 output tile (2x2 WMMA tiles, full fragment reuse);
// grid = (ceil(M/32), ceil(N/256)). OOB rows/cols are address-clamped (their D lanes never stored).
__global__ void __launch_bounds__(256)
wmma_gemm(const float* __restrict__ A, int lda,
          const __bf16* __restrict__ Bw, int ldb,
          float* __restrict__ C, int ldc,
          const float* __restrict__ Cadd, int ldadd,
          int M, int N, int K) {
  const int lane = threadIdx.x & 31;
  const int wave = threadIdx.x >> 5;
  const int lr   = lane & 15;
  const int kw   = (lane >> 4) << 3;   // A K-window
  const int kh   = (lane >> 4) << 4;   // B K-half
  const int m0   = blockIdx.x << 5;
  const int n0   = (blockIdx.y * 8 + wave) << 5;
  if (n0 >= N) return;

  const float* arow0 = A + (size_t)imin(m0 + lr,      M - 1) * lda;
  const float* arow1 = A + (size_t)imin(m0 + 16 + lr, M - 1) * lda;
  const __bf16* brow0 = Bw + (size_t)imin(n0 + lr,      N - 1) * ldb;
  const __bf16* brow1 = Bw + (size_t)imin(n0 + 16 + lr, N - 1) * ldb;

  v8f acc00 = {};
  v8f acc01 = {};
  v8f acc10 = {};
  v8f acc11 = {};
  for (int k = 0; k < K; k += 32) {
    v16bf a0 = load_a_frag_f32(arow0 + k, kw);
    v16bf a1 = load_a_frag_f32(arow1 + k, kw);
    v16bf b0 = load_b_frag(brow0 + k, kh);
    v16bf b1 = load_b_frag(brow1 + k, kh);
    acc00 = __builtin_amdgcn_wmma_f32_16x16x32_bf16(false, a0, false, b0, (short)0, acc00, false, false);
    acc01 = __builtin_amdgcn_wmma_f32_16x16x32_bf16(false, a0, false, b1, (short)0, acc01, false, false);
    acc10 = __builtin_amdgcn_wmma_f32_16x16x32_bf16(false, a1, false, b0, (short)0, acc10, false, false);
    acc11 = __builtin_amdgcn_wmma_f32_16x16x32_bf16(false, a1, false, b1, (short)0, acc11, false, false);
  }

  const int mb = (lane >> 4) << 3;     // D row group within 16-row tile
#pragma unroll
  for (int v = 0; v < 8; ++v) {
    int mA = m0 + mb + v;
    int mB = mA + 16;
    int nA = n0 + lr;
    int nB = n0 + 16 + lr;
    if (mA < M) {
      if (nA < N) {
        float s = acc00[v];
        if (Cadd) s += Cadd[(size_t)mA * ldadd + nA];
        C[(size_t)mA * ldc + nA] = s;
      }
      if (nB < N) {
        float s = acc01[v];
        if (Cadd) s += Cadd[(size_t)mA * ldadd + nB];
        C[(size_t)mA * ldc + nB] = s;
      }
    }
    if (mB < M) {
      if (nA < N) {
        float s = acc10[v];
        if (Cadd) s += Cadd[(size_t)mB * ldadd + nA];
        C[(size_t)mB * ldc + nA] = s;
      }
      if (nB < N) {
        float s = acc11[v];
        if (Cadd) s += Cadd[(size_t)mB * ldadd + nB];
        C[(size_t)mB * ldc + nB] = s;
      }
    }
  }
}

// ---------------- liquid RNN: persistent single-workgroup WMMA scan ----------------
// h (8x512) in LDS (f32 + bf16 mirror, rows 8..15 zero-padded for the 16-row A tile).
// Per step: hW = h @ WrecT^T via WMMA (16 waves x 32 cols each), then tanh/leak update.
__global__ void __launch_bounds__(512)
liquid_scan_kernel(const float* __restrict__ pre, const __bf16* __restrict__ WrecT,
                   const float* __restrict__ lq_b, const float* __restrict__ lq_tau,
                   float* __restrict__ xout) {
  const int HBS = DM + 8;              // padded bf16 row stride (banks)
  __shared__ float  hf[BATCH * DM];
  __shared__ __bf16 hb[16 * (DM + 8)];
  __shared__ float  rtau[DM];
  __shared__ float  bias[DM];

  const int tid = threadIdx.x;
  for (int i = tid; i < BATCH * DM; i += 512) hf[i] = 0.0f;
  for (int i = tid; i < 16 * HBS; i += 512)  hb[i] = (__bf16)0.0f;
  {
    float tv = lq_tau[tid];
    float sp = (tv > 20.0f) ? tv : log1pf(expf(tv));
    rtau[tid] = 1.0f / (sp + 0.1f);
    bias[tid] = lq_b[tid];
  }
  __syncthreads();

  const int lane = tid & 31;
  const int wave = tid >> 5;
  const int lr   = lane & 15;
  const int kw   = (lane >> 4) << 3;
  const int kh   = (lane >> 4) << 4;
  const __bf16* hrow = hb + lr * HBS;
  const __bf16* br0  = WrecT + (size_t)(wave * 32 + lr) * DM;
  const __bf16* br1  = WrecT + (size_t)(wave * 32 + 16 + lr) * DM;

  for (int t = 0; t < SEQ; ++t) {
    v8f acc0 = {};
    v8f acc1 = {};
    for (int k = 0; k < DM; k += 32) {
      v16bf a  = load_a_frag_bf16(hrow + k, kw);
      v16bf b0 = load_b_frag(br0 + k, kh);
      v16bf b1 = load_b_frag(br1 + k, kh);
      acc0 = __builtin_amdgcn_wmma_f32_16x16x32_bf16(false, a, false, b0, (short)0, acc0, false, false);
      acc1 = __builtin_amdgcn_wmma_f32_16x16x32_bf16(false, a, false, b1, (short)0, acc1, false, false);
    }
    __syncthreads();                   // all LDS reads of h done
    if (lane < 16) {                   // D rows 0..7 = batches (rows 8..15 are pad)
      int c0 = wave * 32 + lr;
      int c1 = c0 + 16;
#pragma unroll
      for (int v = 0; v < 8; ++v) {    // v = batch index
        size_t tok = (size_t)(v * SEQ + t);
        float p0 = pre[tok * DM + c0];
        float a0 = tanhf(p0 + acc0[v] + bias[c0]);
        float h0 = hf[v * DM + c0];
        h0 += (a0 - h0) * rtau[c0];
        hf[v * DM + c0] = h0;
        hb[v * HBS + c0] = (__bf16)h0;
        xout[tok * DM + c0] = h0;

        float p1 = pre[tok * DM + c1];
        float a1 = tanhf(p1 + acc1[v] + bias[c1]);
        float h1 = hf[v * DM + c1];
        h1 += (a1 - h1) * rtau[c1];
        hf[v * DM + c1] = h1;
        hb[v * HBS + c1] = (__bf16)h1;
        xout[tok * DM + c1] = h1;
      }
    }
    __syncthreads();                   // h updated before next step's WMMA
  }
}

// ---------------- elementwise / support kernels ----------------
__global__ void cvt_bf16(const float* __restrict__ src, __bf16* __restrict__ dst, size_t n) {
  size_t i = (size_t)blockIdx.x * blockDim.x + threadIdx.x;
  size_t st = (size_t)gridDim.x * blockDim.x;
  for (; i < n; i += st) dst[i] = (__bf16)src[i];
}

// src (rows, cols) f32 -> dst (cols, rows) bf16
__global__ void cvt_bf16_t(const float* __restrict__ src, __bf16* __restrict__ dst, int rows, int cols) {
  int i = blockIdx.x * blockDim.x + threadIdx.x;
  if (i >= rows * cols) return;
  int r = i / cols, c = i - r * cols;
  dst[(size_t)c * rows + r] = (__bf16)src[i];
}

__global__ void layernorm_kernel(const float* __restrict__ X, const float* __restrict__ g,
                                 const float* __restrict__ b, float* __restrict__ Y, int tokens) {
  int wv = blockIdx.x * (blockDim.x >> 5) + (threadIdx.x >> 5);
  int lane = threadIdx.x & 31;
  if (wv >= tokens) return;
  const float* row = X + (size_t)wv * DM;
  float s = 0.0f, ss = 0.0f;
  for (int i = lane; i < DM; i += 32) { float v = row[i]; s += v; ss += v * v; }
#pragma unroll
  for (int off = 16; off > 0; off >>= 1) { s += __shfl_xor(s, off, 32); ss += __shfl_xor(ss, off, 32); }
  float mean = s * (1.0f / DM);
  float var  = ss * (1.0f / DM) - mean * mean;
  float rstd = rsqrtf(var + 1e-5f);
  float* o = Y + (size_t)wv * DM;
  for (int i = lane; i < DM; i += 32) o[i] = (row[i] - mean) * rstd * g[i] + b[i];
}

// causal depthwise conv (k=4) over xm half of xz, + bias, silu -> XM (TOK, DI)
__global__ void conv_silu_kernel(const float* __restrict__ xz, const float* __restrict__ cw,
                                 const float* __restrict__ cb, float* __restrict__ xm) {
  int idx = blockIdx.x * blockDim.x + threadIdx.x;      // t*DI + d
  int t = idx >> 10, d = idx & (DI - 1);
  int b = t >> 9, s = t & (SEQ - 1);
  float acc = cb[d];
#pragma unroll
  for (int kk = 0; kk < 4; ++kk) {
    int sp = s - 3 + kk;
    if (sp >= 0) acc += xz[((size_t)(b * SEQ + sp)) * (2 * DI) + d] * cw[d * 4 + kk];
  }
  float sig = 1.0f / (1.0f + expf(-acc));
  xm[idx] = acc * sig;
}

__global__ void softplus_bias_kernel(float* __restrict__ dt, const float* __restrict__ db) {
  int idx = blockIdx.x * blockDim.x + threadIdx.x;
  int d = idx & (DI - 1);
  float v = dt[idx] + db[d];
  dt[idx] = (v > 20.0f) ? v : log1pf(expf(v));
}

__global__ void silu_kernel(const float* __restrict__ x, float* __restrict__ y, int n) {
  int i = blockIdx.x * blockDim.x + threadIdx.x;
  if (i >= n) return;
  float v = x[i];
  y[i] = v / (1.0f + expf(-v));
}

// Cox-de Boor cubic B-spline basis on fixed grid [-2.2, 2.2], h=0.4 -> 8 coeffs
__device__ __forceinline__ void bspline8(float x, float* out) {
  const float g0 = -2.2f, hs = 0.4f;
  float b[11];
#pragma unroll
  for (int j = 0; j < 11; ++j) {
    float gl = g0 + hs * j;
    b[j] = (x >= gl && x < gl + hs) ? 1.0f : 0.0f;
  }
#pragma unroll
  for (int p = 1; p <= 3; ++p) {
    float inv = 1.0f / (hs * p);
#pragma unroll
    for (int j = 0; j < 11 - p; ++j) {
      float gl  = g0 + hs * j;
      float grp = gl + hs * p;
      b[j] = (x - gl) * inv * b[j] + ((grp + hs) - x) * inv * b[j + 1];
    }
  }
#pragma unroll
  for (int j = 0; j < 8; ++j) out[j] = b[j];
}

// expand X (tokens, DM) -> S (tokens, DM*8) flattened (i*8+k), matching spline_w reshape
__global__ void spline_kernel(const float* __restrict__ X, float* __restrict__ S, int tokens) {
  int idx = blockIdx.x * blockDim.x + threadIdx.x;      // t*DM + i
  if (idx >= tokens * DM) return;
  float c[8];
  bspline8(X[idx], c);
  F4* dst = (F4*)(S + (size_t)idx * 8);
  F4 lo; lo.x = c[0]; lo.y = c[1]; lo.z = c[2]; lo.w = c[3];
  F4 hi; hi.x = c[4]; hi.y = c[5]; hi.z = c[6]; hi.w = c[7];
  dst[0] = lo; dst[1] = hi;
}

// selective scan: thread = (batch b, channel d), 16 states in registers, loop over S
__global__ void mamba_scan_kernel(const float* __restrict__ dtp, const float* __restrict__ xdbl,
                                  const float* __restrict__ u, const float* __restrict__ xz,
                                  const float* __restrict__ alog, const float* __restrict__ Dp,
                                  float* __restrict__ y) {
  int d = blockIdx.x * blockDim.x + threadIdx.x;   // 0..DI-1
  int b = blockIdx.y;
  float a[DST], h[DST];
#pragma unroll
  for (int n = 0; n < DST; ++n) { a[n] = -expf(alog[d * DST + n]); h[n] = 0.0f; }
  float dp = Dp[d];
  for (int s = 0; s < SEQ; ++s) {
    size_t t = (size_t)b * SEQ + s;
    float dtv = dtp[t * DI + d];
    float uv  = u[t * DI + d];
    const float* bc = xdbl + t * 64 + DTR;
    const float* cc = xdbl + t * 64 + DTR + DST;
    float yv = 0.0f;
#pragma unroll
    for (int n = 0; n < DST; ++n) {
      h[n] = expf(dtv * a[n]) * h[n] + dtv * bc[n] * uv;
      yv += h[n] * cc[n];
    }
    float zv = xz[t * (2 * DI) + DI + d];
    float sig = 1.0f / (1.0f + expf(-zv));
    y[t * DI + d] = (yv + dp * uv) * (zv * sig);
  }
}

// mean over sequence: X (B,S,DM) -> out (B,DM)
__global__ void mean_kernel(const float* __restrict__ X, float* __restrict__ out) {
  int idx = blockIdx.x * blockDim.x + threadIdx.x;   // b*DM + d
  if (idx >= BATCH * DM) return;
  int b = idx >> 9, d = idx & (DM - 1);
  const float* p = X + ((size_t)b * SEQ) * DM + d;
  float s = 0.0f;
  for (int t = 0; t < SEQ; ++t) s += p[(size_t)t * DM];
  out[idx] = s * (1.0f / SEQ);
}

// ---------------- workspace layout (bytes) ----------------
static constexpr size_t SZ_TOKDM = (size_t)TOK * DM * 4;   // 8 MB
static constexpr size_t SZ_TOKDI = (size_t)TOK * DI * 4;   // 16 MB
static constexpr size_t OFF_WINT   = 0;
static constexpr size_t OFF_WRECT  = OFF_WINT   + (size_t)DM * DIN * 2;
static constexpr size_t OFF_INPROJ = OFF_WRECT  + (size_t)DM * DM * 2;
static constexpr size_t OFF_XPROJ  = OFF_INPROJ + (size_t)NL * 2048 * 512 * 2;
static constexpr size_t OFF_DTPROJ = OFF_XPROJ  + (size_t)NL * 64 * 1024 * 2;
static constexpr size_t OFF_OUTPROJ= OFF_DTPROJ + (size_t)NL * 1024 * 32 * 2;
static constexpr size_t OFF_KBASE  = OFF_OUTPROJ+ (size_t)NL * 512 * 1024 * 2;
static constexpr size_t OFF_KSPL   = OFF_KBASE  + (size_t)NL * 512 * 512 * 2;
static constexpr size_t OFF_CBASE  = OFF_KSPL   + (size_t)NL * 512 * 4096 * 2;
static constexpr size_t OFF_CSPL   = OFF_CBASE  + (size_t)16 * 512 * 2;
static constexpr size_t OFF_PRE    = OFF_CSPL   + (size_t)16 * 4096 * 2;
static constexpr size_t OFF_X0     = OFF_PRE  + SZ_TOKDM;
static constexpr size_t OFF_X1     = OFF_X0   + SZ_TOKDM;
static constexpr size_t OFF_XN     = OFF_X1   + SZ_TOKDM;
static constexpr size_t OFF_XZ     = OFF_XN   + SZ_TOKDM;
static constexpr size_t OFF_XM     = OFF_XZ   + (size_t)TOK * 2048 * 4;
static constexpr size_t OFF_DT     = OFF_XM   + SZ_TOKDI;
static constexpr size_t OFF_XDBL   = OFF_DT   + SZ_TOKDI;
static constexpr size_t OFF_Y      = OFF_XDBL + (size_t)TOK * 64 * 4;
static constexpr size_t OFF_XM2    = OFF_Y    + SZ_TOKDI;
static constexpr size_t OFF_SILU   = OFF_XM2  + SZ_TOKDM;
static constexpr size_t OFF_XB     = OFF_SILU + SZ_TOKDM;
static constexpr size_t OFF_SILUB  = OFF_XB   + (size_t)BATCH * DM * 4;
static constexpr size_t OFF_SPLB   = OFF_SILUB+ (size_t)BATCH * DM * 4;
// 64MB spline matrix aliases XZ+XM+DT (all dead when it is produced)
static constexpr size_t OFF_SPL    = OFF_XZ;

extern "C" void kernel_launch(void* const* d_in, const int* in_sizes, int n_in,
                              void* d_out, int out_size, void* d_ws, size_t ws_size,
                              hipStream_t stream) {
  (void)in_sizes; (void)n_in; (void)out_size; (void)ws_size;
  const float* x        = (const float*)d_in[0];
  const float* lq_Win   = (const float*)d_in[1];
  const float* lq_Wrec  = (const float*)d_in[2];
  const float* lq_b     = (const float*)d_in[3];
  const float* lq_tau   = (const float*)d_in[4];
  const float* ln_g     = (const float*)d_in[5];
  const float* ln_bb    = (const float*)d_in[6];
  const float* in_proj  = (const float*)d_in[7];
  const float* conv_w   = (const float*)d_in[8];
  const float* conv_b   = (const float*)d_in[9];
  const float* x_proj   = (const float*)d_in[10];
  const float* dt_proj  = (const float*)d_in[11];
  const float* dt_pb    = (const float*)d_in[12];
  const float* A_log    = (const float*)d_in[13];
  const float* Dp       = (const float*)d_in[14];
  const float* out_proj = (const float*)d_in[15];
  const float* kbase    = (const float*)d_in[16];
  const float* kspl     = (const float*)d_in[17];
  const float* fn_g     = (const float*)d_in[18];
  const float* fn_b     = (const float*)d_in[19];
  const float* cbase    = (const float*)d_in[20];
  const float* cspl     = (const float*)d_in[21];
  float* out = (float*)d_out;

  char* ws = (char*)d_ws;
  __bf16* WINT    = (__bf16*)(ws + OFF_WINT);
  __bf16* WRECT   = (__bf16*)(ws + OFF_WRECT);
  __bf16* INPROJ  = (__bf16*)(ws + OFF_INPROJ);
  __bf16* XPROJ   = (__bf16*)(ws + OFF_XPROJ);
  __bf16* DTPROJ  = (__bf16*)(ws + OFF_DTPROJ);
  __bf16* OUTPROJ = (__bf16*)(ws + OFF_OUTPROJ);
  __bf16* KBASE   = (__bf16*)(ws + OFF_KBASE);
  __bf16* KSPL    = (__bf16*)(ws + OFF_KSPL);
  __bf16* CBASE   = (__bf16*)(ws + OFF_CBASE);
  __bf16* CSPL    = (__bf16*)(ws + OFF_CSPL);
  float* PRE  = (float*)(ws + OFF_PRE);
  float* X0   = (float*)(ws + OFF_X0);
  float* X1   = (float*)(ws + OFF_X1);
  float* XN   = (float*)(ws + OFF_XN);
  float* XZ   = (float*)(ws + OFF_XZ);
  float* XM   = (float*)(ws + OFF_XM);
  float* DTB  = (float*)(ws + OFF_DT);
  float* XDBL = (float*)(ws + OFF_XDBL);
  float* Y    = (float*)(ws + OFF_Y);
  float* XM2  = (float*)(ws + OFF_XM2);
  float* SILU = (float*)(ws + OFF_SILU);
  float* XB   = (float*)(ws + OFF_XB);
  float* SILB = (float*)(ws + OFF_SILUB);
  float* SPLB = (float*)(ws + OFF_SPLB);
  float* SPL  = (float*)(ws + OFF_SPL);

  auto cvt = [&](const float* s, __bf16* d, size_t n) {
    int g = (int)((n + 255) / 256); if (g > 16384) g = 16384;
    cvt_bf16<<<g, 256, 0, stream>>>(s, d, n);
  };
  auto gemm = [&](const float* A, int lda, const __bf16* B, int ldb, float* C, int ldc,
                  const float* Cadd, int ldadd, int M, int N, int K) {
    dim3 g((M + 31) / 32, (N + 255) / 256);
    wmma_gemm<<<g, 256, 0, stream>>>(A, lda, B, ldb, C, ldc, Cadd, ldadd, M, N, K);
  };

  // ---- weight conversion to bf16 (and transposes for the liquid path) ----
  cvt_bf16_t<<<(DIN * DM + 255) / 256, 256, 0, stream>>>(lq_Win, WINT, DIN, DM);   // (512,64)
  cvt_bf16_t<<<(DM * DM + 255) / 256, 256, 0, stream>>>(lq_Wrec, WRECT, DM, DM);   // (512,512)
  cvt(in_proj,  INPROJ,  (size_t)NL * 2048 * 512);
  cvt(x_proj,   XPROJ,   (size_t)NL * 64 * 1024);
  cvt(dt_proj,  DTPROJ,  (size_t)NL * 1024 * 32);
  cvt(out_proj, OUTPROJ, (size_t)NL * 512 * 1024);
  cvt(kbase,    KBASE,   (size_t)NL * 512 * 512);
  cvt(kspl,     KSPL,    (size_t)NL * 512 * 4096);
  cvt(cbase,    CBASE,   (size_t)10 * 512);
  cvt(cspl,     CSPL,    (size_t)10 * 4096);

  // ---- liquid projection + sequential liquid scan ----
  gemm(x, DIN, WINT, DIN, PRE, DM, nullptr, 0, TOK, DM, DIN);
  liquid_scan_kernel<<<1, 512, 0, stream>>>(PRE, WRECT, lq_b, lq_tau, X0);

  // ---- NUM_LAYERS mamba+KAN blocks ----
  float* Xc = X0;
  float* Xn = X1;
  for (int l = 0; l < NL; ++l) {
    layernorm_kernel<<<TOK / 8, 256, 0, stream>>>(Xc, ln_g + l * DM, ln_bb + l * DM, XN, TOK);
    gemm(XN, DM, INPROJ + (size_t)l * 2048 * 512, 512, XZ, 2048, nullptr, 0, TOK, 2048, 512);
    conv_silu_kernel<<<(TOK * DI) / 256, 256, 0, stream>>>(XZ, conv_w + l * DI * 4, conv_b + l * DI, XM);
    gemm(XM, DI, XPROJ + (size_t)l * 64 * 1024, 1024, XDBL, 64, nullptr, 0, TOK, 64, DI);
    gemm(XDBL, 64, DTPROJ + (size_t)l * 1024 * 32, 32, DTB, DI, nullptr, 0, TOK, DI, DTR);
    softplus_bias_kernel<<<(TOK * DI) / 256, 256, 0, stream>>>(DTB, dt_pb + l * DI);
    mamba_scan_kernel<<<dim3(DI / 256, BATCH), 256, 0, stream>>>(
        DTB, XDBL, XM, XZ, A_log + (size_t)l * DI * DST, Dp + l * DI, Y);
    gemm(Y, DI, OUTPROJ + (size_t)l * 512 * 1024, 1024, XM2, DM, nullptr, 0, TOK, DM, DI);
    // KAN: out = residual + silu(xm2)@base^T + splines(xm2)@spline^T
    silu_kernel<<<(TOK * DM) / 256, 256, 0, stream>>>(XM2, SILU, TOK * DM);
    spline_kernel<<<(TOK * DM) / 256, 256, 0, stream>>>(XM2, SPL, TOK);   // overwrites XZ/XM/DT (dead)
    gemm(SILU, DM, KBASE + (size_t)l * 512 * 512, 512, Xn, DM, Xc, DM, TOK, DM, DM);
    gemm(SPL, 4096, KSPL + (size_t)l * 512 * 4096, 4096, Xn, DM, Xn, DM, TOK, DM, 4096);
    float* t = Xc; Xc = Xn; Xn = t;
  }

  // ---- final norm, sequence mean, classifier KAN ----
  layernorm_kernel<<<TOK / 8, 256, 0, stream>>>(Xc, fn_g, fn_b, XN, TOK);
  mean_kernel<<<(BATCH * DM) / 256, 256, 0, stream>>>(XN, XB);
  silu_kernel<<<(BATCH * DM) / 256, 256, 0, stream>>>(XB, SILB, BATCH * DM);
  spline_kernel<<<(BATCH * DM) / 256, 256, 0, stream>>>(XB, SPLB, BATCH);
  gemm(SILB, DM, CBASE, 512, out, 10, nullptr, 0, BATCH, 10, DM);
  gemm(SPLB, 4096, CSPL, 4096, out, 10, out, 10, BATCH, 10, 4096);
}
